// AUSTGN_21492016349427
// MI455X (gfx1250) — matrix operand order repaired and checked by reading, not compiled, not run on hardware
//
#include <hip/hip_runtime.h>
#include <hip/hip_bf16.h>

// ---------------------------------------------------------------------------
// AUSTGN for MI455X (gfx1250): bf16 WMMA GEMMs with double-buffered
// async-to-LDS staging + persistent wave32 recurrent scan with
// register-resident cell state.
// ---------------------------------------------------------------------------

#define B_    256
#define T_    512
#define IPT_  512
#define HID_  512
#define M_    (B_ * T_)          // 131072 rows of x

typedef __bf16 bf16;
typedef __attribute__((ext_vector_type(16))) __bf16 v16bf;
typedef __attribute__((ext_vector_type(8)))  __bf16 v8bf;
typedef __attribute__((ext_vector_type(8)))  float  v8f;

__device__ __forceinline__ float sigf(float x) { return 1.0f / (1.0f + __expf(-x)); }

// LDS byte address for async-to-LDS: addrspacecast generic->AS(3) then
// ptrtoint. This both produces the raw LDS offset the hardware wants AND
// escapes the LDS object into the asm, so the compiler cannot delete the
// ds_load reads of data written by the async DMA.
__device__ __forceinline__ unsigned lds_addr_u32(const void* p) {
  return (unsigned)(unsigned long long)(__attribute__((address_space(3))) const char*)p;
}

// Concatenate two 8x bf16 chunks into one 16x fragment (maps to the WMMA
// per-lane VGPR layout: chunk0 -> VGPRs 0-3, chunk1 -> VGPRs 4-7).
__device__ __forceinline__ v16bf frag_cat(const bf16* p0, const bf16* p1) {
  v8bf lo = *(const v8bf*)p0;
  v8bf hi = *(const v8bf*)p1;
  v16bf r;
#pragma unroll
  for (int i = 0; i < 8; ++i) { r[i] = lo[i]; r[i + 8] = hi[i]; }
  return r;
}

// ---------------------------------------------------------------------------
// Kernel 1: split input (B,T,2+IPT) f32 -> Tt, Dt (f32) and contiguous x bf16
// ---------------------------------------------------------------------------
__global__ void cvt_input(const float* __restrict__ in, bf16* __restrict__ xbf,
                          float* __restrict__ Tt, float* __restrict__ Dt) {
  long total = (long)M_ * (2 + IPT_);
  for (long i = (long)blockIdx.x * blockDim.x + threadIdx.x; i < total;
       i += (long)gridDim.x * blockDim.x) {
    long row = i / (2 + IPT_);
    int  col = (int)(i - row * (2 + IPT_));
    float v = in[i];
    if (col == 0)       Tt[row] = v;
    else if (col == 1)  Dt[row] = v;
    else                xbf[row * IPT_ + (col - 2)] = (bf16)v;
  }
}

// ---------------------------------------------------------------------------
// Kernel 2: W (K=512 x N=512, f32, row-major K) -> WT (N x K, bf16) so the
// WMMA B-operand (column of W) is a contiguous-K row of WT.
// ---------------------------------------------------------------------------
__global__ void cvt_wT(const float* __restrict__ W, bf16* __restrict__ WT) {
  int n = blockIdx.x;                         // 0..511
  for (int k = threadIdx.x; k < HID_; k += blockDim.x)
    WT[(long)n * HID_ + k] = (bf16)W[(long)k * HID_ + n];
}

// ---------------------------------------------------------------------------
// Kernel 3: raw[g] = x @ Wx_g   (M=131072, K=512, N=512), 8 gates via grid.z.
// 128x128 block tile, 8 wave32 waves arranged 4x2, each wave 32x64 (2x4
// subtiles of 16x16). A tile double-buffered in LDS via async DMA: issue
// tile k+1, compute tile k, two workgroup barriers per K step.
// ---------------------------------------------------------------------------
struct GemmArgs { const bf16* w[8]; float* o[8]; };

__global__ void __launch_bounds__(256)
gemm_xw(const bf16* __restrict__ X, GemmArgs p) {
  __shared__ __align__(128) bf16 As[2][128 * 32];  // 2 x 8 KB double buffer

  const int g     = blockIdx.z;
  const bf16* WT  = p.w[g];
  float*      out = p.o[g];
  const int mBase = blockIdx.x * 128;
  const int nBase = blockIdx.y * 128;

  const int wave   = threadIdx.x >> 5;
  const int lane   = threadIdx.x & 31;
  const int laneLo = lane & 15;
  const int laneHi = lane >> 4;
  const int wr = wave >> 1;                        // 0..3  (M strip of 32)
  const int wc = wave & 1;                         // 0..1  (N strip of 64)
  const int nW = nBase + wc * 64;

  v8f acc[2][4] = {};

  // Issue async DMA for one 128x32 bf16 A tile (8 KB): 2 x 16 B per thread.
  auto issue_tile = [&](int k0, int buf) {
#pragma unroll
    for (int ch = 0; ch < 2; ++ch) {
      int idx = ((int)threadIdx.x * 2 + ch) * 8;   // element index into As[buf]
      int row = idx >> 5;
      int kk  = idx & 31;
      unsigned laddr = lds_addr_u32(&As[buf][idx]);
      unsigned long long gaddr =
          (unsigned long long)(X + (long)(mBase + row) * IPT_ + k0 + kk);
      asm volatile("global_load_async_to_lds_b128 %0, %1, off"
                   :: "v"(laddr), "v"(gaddr) : "memory");
    }
  };

  issue_tile(0, 0);

  for (int it = 0; it < IPT_ / 32; ++it) {
    const int k0 = it * 32;
    asm volatile("s_wait_asynccnt 0" ::: "memory"); // tile `it` landed in LDS
    __syncthreads();  // all waves: tile visible; buffer (it+1)&1 free to refill
    if (it + 1 < IPT_ / 32) issue_tile(k0 + 32, (it + 1) & 1);

    const bf16* Ab = As[it & 1];
#pragma unroll
    for (int ms = 0; ms < 2; ++ms) {
      const bf16* ab = &Ab[(wr * 32 + ms * 16 + laneLo) * 32 + laneHi * 8];
      v16bf a = frag_cat(ab, ab + 16);
#pragma unroll
      for (int ns = 0; ns < 4; ++ns) {
        const bf16* bb = WT + (long)(nW + ns * 16 + laneLo) * HID_ + k0 + laneHi * 8;
        v16bf b = frag_cat(bb, bb + 16);
        acc[ms][ns] = __builtin_amdgcn_wmma_f32_16x16x32_bf16(
            false, a, false, b, (short)0, acc[ms][ns], false, false);
      }
    }
    __syncthreads();  // everyone done reading As[it&1] before it is refilled
  }

#pragma unroll
  for (int ms = 0; ms < 2; ++ms)
#pragma unroll
    for (int ns = 0; ns < 4; ++ns)
#pragma unroll
      for (int r = 0; r < 8; ++r) {
        int mm = mBase + wr * 32 + ms * 16 + r + laneHi * 8;
        int nn = nW + ns * 16 + laneLo;
        out[(long)mm * HID_ + nn] = acc[ms][ns][r];
      }
}

// ---------------------------------------------------------------------------
// Kernel 4: fuse biases / Tt / Dt terms and the td1,td2 sigmoid products.
// ---------------------------------------------------------------------------
__global__ void epilogue(float* r_i, float* r_f, float* r_c, float* r_o,
                         float* r_t1, float* r_t2,
                         const float* __restrict__ r_d1, const float* __restrict__ r_d2,
                         const float* __restrict__ Tt, const float* __restrict__ Dt,
                         const float* __restrict__ bi, const float* __restrict__ bff,
                         const float* __restrict__ bc,
                         const float* __restrict__ Wto, const float* __restrict__ Wdo,
                         const float* __restrict__ bo,
                         const float* __restrict__ Wt1, const float* __restrict__ bt1,
                         const float* __restrict__ Wt2, const float* __restrict__ bt2,
                         const float* __restrict__ Wd1, const float* __restrict__ bd1,
                         const float* __restrict__ Wd2, const float* __restrict__ bd2) {
  long total = (long)M_ * HID_;
  for (long i = (long)blockIdx.x * blockDim.x + threadIdx.x; i < total;
       i += (long)gridDim.x * blockDim.x) {
    long row = i >> 9;
    int  n   = (int)(i & (HID_ - 1));
    float tt = Tt[row], dt = Dt[row];
    r_i[i] += bi[n];
    r_f[i] += bff[n];
    r_c[i] += bc[n];
    r_o[i] += tt * Wto[n] + dt * Wdo[n] + bo[n];
    float g1 = sigf(r_t1[i] + sigf(tt * Wt1[n]) + bt1[n]) *
               sigf(r_d1[i] + sigf(dt * Wd1[n]) + bd1[n]);
    float g2 = sigf(r_t2[i] + sigf(tt * Wt2[n]) + bt2[n]) *
               sigf(r_d2[i] + sigf(dt * Wd2[n]) + bd2[n]);
    r_t1[i] = g1;
    r_t2[i] = g2;
  }
}

// ---------------------------------------------------------------------------
// Kernel 5: persistent recurrent scan. One block per 16 batch rows; 8 wave32
// waves each own 64 output columns. h lives in LDS as bf16, c in registers in
// the WMMA C/D lane layout. Per step: 4 gate GEMMs (16x512x512) via
// v_wmma_f32_16x16x32_bf16, then the fused LSTM nonlinearity.
// ---------------------------------------------------------------------------
__global__ void __launch_bounds__(256)
scan(const float* __restrict__ pi, const float* __restrict__ pf,
     const float* __restrict__ pc, const float* __restrict__ po,
     const float* __restrict__ g1, const float* __restrict__ g2,
     const bf16* __restrict__ whT /* [4][512][512] n-major */,
     float* __restrict__ out /* hidden_seq | h_t | c_t */) {
  __shared__ __align__(128) bf16 hbuf[16 * HID_];  // 16 KB

  const int bb     = blockIdx.x * 16;              // batch slice base
  const int wave   = threadIdx.x >> 5;
  const int lane   = threadIdx.x & 31;
  const int laneLo = lane & 15;
  const int laneHi = lane >> 4;
  const int nW     = wave * 64;

  for (int idx = threadIdx.x; idx < 16 * HID_; idx += blockDim.x)
    hbuf[idx] = (bf16)0.0f;

  v8f cst[4] = {};                                 // c state, per N-subtile
  __syncthreads();

  for (int t = 0; t < T_; ++t) {
    v8f acc[4][4] = {};                            // [gate][N-subtile]

    for (int k0 = 0; k0 < HID_; k0 += 32) {
      const bf16* ab = &hbuf[laneLo * HID_ + k0 + laneHi * 8];
      v16bf a = frag_cat(ab, ab + 16);
#pragma unroll
      for (int gg = 0; gg < 4; ++gg) {
#pragma unroll
        for (int ns = 0; ns < 4; ++ns) {
          const bf16* bb2 =
              whT + ((long)gg * HID_ + nW + ns * 16 + laneLo) * HID_ + k0 + laneHi * 8;
          v16bf b = frag_cat(bb2, bb2 + 16);
          acc[gg][ns] = __builtin_amdgcn_wmma_f32_16x16x32_bf16(
              false, a, false, b, (short)0, acc[gg][ns], false, false);
        }
      }
    }

    __syncthreads();                               // all waves done reading hbuf

#pragma unroll
    for (int ns = 0; ns < 4; ++ns) {
      int col = nW + ns * 16 + laneLo;
#pragma unroll
      for (int r = 0; r < 8; ++r) {
        int  b   = bb + r + laneHi * 8;
        long idx = ((long)b * T_ + t) * HID_ + col;
        float it = sigf(pi[idx] + acc[0][ns][r]);
        float ft = sigf(pf[idx] + acc[1][ns][r]);
        float j  = tanhf(pc[idx] + acc[2][ns][r]);
        float ot = sigf(po[idx] + acc[3][ns][r]);
        float cold = cst[ns][r];
        float ij   = it * j;
        float chat = ft * cold + ij * g1[idx];
        float cnew = ft * cold + ij * g2[idx];
        cst[ns][r] = cnew;
        float h = ot * tanhf(chat);
        out[idx] = h;
        hbuf[(r + laneHi * 8) * HID_ + col] = (bf16)h;
        if (t == T_ - 1) {
          out[(long)M_ * HID_ + (long)b * HID_ + col] = h;                      // h_t
          out[(long)M_ * HID_ + (long)B_ * HID_ + (long)b * HID_ + col] = cnew; // c_t
        }
      }
    }
    __syncthreads();                               // hbuf updated for next step
  }
}

// ---------------------------------------------------------------------------
extern "C" void kernel_launch(void* const* d_in, const int* in_sizes, int n_in,
                              void* d_out, int out_size, void* d_ws, size_t ws_size,
                              hipStream_t stream) {
  const float* input = (const float*)d_in[0];
  const float* Wxi = (const float*)d_in[3],  *Whi = (const float*)d_in[4],  *bi  = (const float*)d_in[5];
  const float* Wxf = (const float*)d_in[6],  *Whf = (const float*)d_in[7],  *bf_ = (const float*)d_in[8];
  const float* Wxc = (const float*)d_in[9],  *Whc = (const float*)d_in[10], *bc  = (const float*)d_in[11];
  const float* Wxt1= (const float*)d_in[12], *Wt1 = (const float*)d_in[13], *bt1 = (const float*)d_in[14];
  const float* Wxt2= (const float*)d_in[15], *Wt2 = (const float*)d_in[16], *bt2 = (const float*)d_in[17];
  const float* Wxd1= (const float*)d_in[18], *Wd1 = (const float*)d_in[19], *bd1 = (const float*)d_in[20];
  const float* Wxd2= (const float*)d_in[21], *Wd2 = (const float*)d_in[22], *bd2 = (const float*)d_in[23];
  const float* Wxo = (const float*)d_in[24], *Who = (const float*)d_in[25];
  const float* Wto = (const float*)d_in[26], *Wdo = (const float*)d_in[27], *bo  = (const float*)d_in[28];
  float* out = (float*)d_out;

  // Workspace carve-up
  char* ws = (char*)d_ws;
  size_t off = 0;
  bf16* xbf = (bf16*)(ws + off);  off += (size_t)M_ * IPT_ * sizeof(bf16);
  float* Tt = (float*)(ws + off); off += (size_t)M_ * sizeof(float);
  float* Dt = (float*)(ws + off); off += (size_t)M_ * sizeof(float);
  bf16* wxT = (bf16*)(ws + off);  off += (size_t)8 * IPT_ * HID_ * sizeof(bf16);
  bf16* whT = (bf16*)(ws + off);  off += (size_t)4 * HID_ * HID_ * sizeof(bf16);
  float* raw[8];
  for (int g = 0; g < 8; ++g) { raw[g] = (float*)(ws + off); off += (size_t)M_ * HID_ * sizeof(float); }

  // 1. input split + bf16 conversion
  cvt_input<<<8192, 256, 0, stream>>>(input, xbf, Tt, Dt);

  // 2. weight transpose + bf16 conversion
  const float* wx[8] = { Wxi, Wxf, Wxc, Wxo, Wxt1, Wxt2, Wxd1, Wxd2 };
  for (int g = 0; g < 8; ++g)
    cvt_wT<<<HID_, 256, 0, stream>>>(wx[g], wxT + (size_t)g * IPT_ * HID_);
  const float* wh[4] = { Whi, Whf, Whc, Who };
  for (int g = 0; g < 4; ++g)
    cvt_wT<<<HID_, 256, 0, stream>>>(wh[g], whT + (size_t)g * HID_ * HID_);

  // 3. eight big GEMMs: raw[g] = x @ Wx_g
  GemmArgs ga;
  for (int g = 0; g < 8; ++g) { ga.w[g] = wxT + (size_t)g * IPT_ * HID_; ga.o[g] = raw[g]; }
  gemm_xw<<<dim3(M_ / 128, HID_ / 128, 8), 256, 0, stream>>>(xbf, ga);

  // 4. fused gate-prep epilogue (raw0..3 -> pre_i/f/c/o, raw4/5 -> g1/g2)
  epilogue<<<8192, 256, 0, stream>>>(raw[0], raw[1], raw[2], raw[3], raw[4], raw[5],
                                     raw[6], raw[7], Tt, Dt, bi, bf_, bc,
                                     Wto, Wdo, bo, Wt1, bt1, Wt2, bt2, Wd1, bd1, Wd2, bd2);

  // 5. sequential scan (16 persistent blocks, one per 16 batch rows)
  scan<<<B_ / 16, 256, 0, stream>>>(raw[0], raw[1], raw[2], raw[3], raw[4], raw[5],
                                    whT, out);
}
